// MTL_SG_28879360098623
// MI455X (gfx1250) — matrix operand order repaired and checked by reading, not compiled, not run on hardware
//
#include <hip/hip_runtime.h>

#define B_ROWS 16384
#define NEXP 16

typedef __bf16 bf16_t;
typedef bf16_t v16bf __attribute__((ext_vector_type(16)));
typedef float  v8f   __attribute__((ext_vector_type(8)));
typedef float  v4f   __attribute__((ext_vector_type(4)));
typedef unsigned short u16;
typedef u16 v16u __attribute__((ext_vector_type(16)));
typedef u16 v8u  __attribute__((ext_vector_type(8)));
typedef unsigned int uv4 __attribute__((ext_vector_type(4)));
typedef unsigned int uv2 __attribute__((ext_vector_type(2)));

union BF16Frag { v16u u; v16bf b; };
union AFragU   { struct { v8u lo, hi; } h; v16bf b; };

__device__ __forceinline__ u16 f2bf(float f) {
  unsigned u = __float_as_uint(f);
  u += 0x7FFFu + ((u >> 16) & 1u);            // round-to-nearest-even
  return (u16)(u >> 16);
}
__device__ __forceinline__ float bf2f(u16 h) { return __uint_as_float(((unsigned)h) << 16); }
__device__ __forceinline__ void lds_fence() { asm volatile("s_wait_dscnt 0" ::: "memory"); }
__device__ __forceinline__ void async_fence() { asm volatile("s_wait_asynccnt 0" ::: "memory"); }
__device__ __forceinline__ v8f vzero8() { v8f z = {0.f,0.f,0.f,0.f,0.f,0.f,0.f,0.f}; return z; }

__device__ __forceinline__ v8f wmma_bf16(v16bf a, v16bf b, v8f c) {
  return __builtin_amdgcn_wmma_f32_16x16x32_bf16(false, a, false, b, (short)0, c, false, false);
}

// A fragment (16 rows x 32 K, bf16) from LDS row-major [16][ldk].
// Layout (matches B): lanes 0-15 (m=lane) hold K = kb..kb+15;
// lanes 16-31 hold K = kb+16..kb+31.
__device__ __forceinline__ v16bf load_a_frag(const u16* lds, int ldk, int kb, int lane) {
  const int m = lane & 15;
  const int k = kb + ((lane >> 4) << 4);
  const u16* p = lds + m * ldk + k;
  AFragU r;
  r.h.lo = *(const v8u*)(p);
  r.h.hi = *(const v8u*)(p + 8);
  return r.b;
}

// B fragment (32 K x 16 N, bf16) from pre-transposed weights Wt[n][K] (contiguous per lane).
// Contiguous k order (matches A): lanes 0-15 (n=lane) hold K = kb..kb+15; lanes 16-31 hold K = kb+16..kb+31.
__device__ __forceinline__ v16bf load_b_frag(const u16* wt, int ldk, int n0, int kb, int lane) {
  const int n = n0 + (lane & 15);
  const int k = kb + ((lane >> 4) << 4);
  BF16Frag r;
  r.u = *(const v16u*)(wt + n * ldk + k);
  return r.b;
}

// one 16x128 output stripe, K=128: 4 k-steps x 8 n-tiles = 32 wmma
__device__ __forceinline__ void gemm16x128(const u16* Abuf, const u16* Wt, v8f acc[8], int lane) {
  #pragma unroll
  for (int t = 0; t < 8; ++t) acc[t] = vzero8();
  #pragma unroll
  for (int kb = 0; kb < 128; kb += 32) {
    v16bf a = load_a_frag(Abuf, 128, kb, lane);
    #pragma unroll
    for (int t = 0; t < 8; ++t) {
      v16bf b = load_b_frag(Wt, 128, t * 16, kb, lane);
      acc[t] = wmma_bf16(a, b, acc[t]);
    }
  }
}

// async-copy one 128x128 bf16 weight matrix (32 KB) into LDS at byte offset ldsByteOff.
// 2048 x 16B chunks, 256 threads x 8 chunks, tracked by ASYNCcnt.
__device__ __forceinline__ void stage_weights_async(const u16* __restrict__ src,
                                                    unsigned ldsByteOff, int tid) {
  #pragma unroll
  for (int i = 0; i < 8; ++i) {
    int c = tid + 256 * i;
    unsigned la = ldsByteOff + (unsigned)c * 16u;
    const char* ga = (const char*)src + (size_t)c * 16u;
    asm volatile("global_load_async_to_lds_b128 %0, %1, off"
                 :: "v"(la), "v"(ga) : "memory");
  }
}

// ---------------- K0: convert f32 W[k][n] -> bf16 W^T[n][k], batched ----------------
__global__ __launch_bounds__(256) void k_conv_t(const float* __restrict__ src,
                                                u16* __restrict__ dst,
                                                int K, int N, int total) {
  int i = blockIdx.x * 256 + threadIdx.x;
  if (i >= total) return;
  int kn = K * N;
  int b = i / kn;
  int rem = i - b * kn;
  int k = rem / N;
  int n = rem - k * N;
  dst[b * kn + n * K + k] = f2bf(src[i]);
}

// ---------------- K1: shared bottom: gather 4 tables, 512->128 GEMM, relu ----------------
__global__ __launch_bounds__(256) void k_shared_bottom(
    const int* __restrict__ ids0, const int* __restrict__ ids1,
    const int* __restrict__ ids2, const int* __restrict__ ids3,
    const float* __restrict__ emb0, const float* __restrict__ emb1,
    const float* __restrict__ emb2, const float* __restrict__ emb3,
    const u16* __restrict__ projWt, const float* __restrict__ projb,
    u16* __restrict__ shared_bf) {
  extern __shared__ u16 ldsA[];                 // 8 waves x [16][512] bf16
  const int lane = threadIdx.x & 31;
  const int wave = threadIdx.x >> 5;
  const int row0 = (blockIdx.x * 8 + wave) * 16;
  u16* A = ldsA + wave * 16 * 512;

  const int* idsAll[4] = {ids0, ids1, ids2, ids3};
  const float* embAll[4] = {emb0, emb1, emb2, emb3};

  for (int r = 0; r < 16; ++r) {
    #pragma unroll
    for (int t = 0; t < 4; ++t) {
      int id = idsAll[t][row0 + r];
      v4f x = *(const v4f*)(embAll[t] + (size_t)id * 128 + lane * 4);
      uv2 pk;
      pk.x = (unsigned)f2bf(x.x) | ((unsigned)f2bf(x.y) << 16);
      pk.y = (unsigned)f2bf(x.z) | ((unsigned)f2bf(x.w) << 16);
      *(uv2*)(A + r * 512 + t * 128 + lane * 4) = pk;
    }
  }
  lds_fence();

  v8f acc[8];
  #pragma unroll
  for (int t = 0; t < 8; ++t) acc[t] = vzero8();
  for (int kb = 0; kb < 512; kb += 32) {
    v16bf a = load_a_frag(A, 512, kb, lane);
    #pragma unroll
    for (int t = 0; t < 8; ++t) {
      v16bf b = load_b_frag(projWt, 512, t * 16, kb, lane);
      acc[t] = wmma_bf16(a, b, acc[t]);
    }
  }

  const int hh = lane >> 4, nlo = lane & 15;
  #pragma unroll
  for (int t = 0; t < 8; ++t) {
    float bias = projb[t * 16 + nlo];
    #pragma unroll
    for (int v = 0; v < 8; ++v) {
      float x = acc[t][v] + bias;
      x = (x > 0.f) ? x : 0.f;
      shared_bf[(size_t)(row0 + v + 8 * hh) * 128 + t * 16 + nlo] = f2bf(x);
    }
  }
}

// ---------------- K2: scene encoder + gate softmax/top-2 + KL partials ----------------
__global__ __launch_bounds__(256) void k_gate(
    const int* __restrict__ resource_id, const float* __restrict__ realtime,
    const float* __restrict__ userp, const float* __restrict__ ctxf,
    const float* __restrict__ p_hist, const u16* __restrict__ shared_bf,
    const float* __restrict__ res_emb,
    const float* __restrict__ scW1, const float* __restrict__ scb1,
    const float* __restrict__ scW2, const float* __restrict__ scb2,
    const float* __restrict__ gateW, const float* __restrict__ gateb,
    float* __restrict__ g_out, float* __restrict__ kl_partial) {
  extern __shared__ float smf[];                // [256][128] h1 + [256] reduction
  const int b = blockIdx.x * 256 + threadIdx.x;
  float* h1 = smf + threadIdx.x * 128;

  float sx[40];
  {
    int rid = resource_id[b];
    #pragma unroll
    for (int i = 0; i < 32; ++i) sx[i] = res_emb[rid * 32 + i];
    #pragma unroll
    for (int i = 0; i < 8; ++i) sx[32 + i] = realtime[b * 8 + i];
  }
  for (int j = 0; j < 128; ++j) {
    float s = scb1[j];
    #pragma unroll
    for (int k = 0; k < 40; ++k) s += sx[k] * scW1[k * 128 + j];
    h1[j] = (s > 0.f) ? s : 0.f;
  }
  lds_fence();

  float logits[16];
  #pragma unroll
  for (int e = 0; e < 16; ++e) logits[e] = gateb[e];
  for (int j = 0; j < 128; ++j) {               // scene rows 0..127 of gate_W
    float s = scb2[j];
    for (int k = 0; k < 128; ++k) s += h1[k] * scW2[k * 128 + j];
    s = (s > 0.f) ? s : 0.f;
    #pragma unroll
    for (int e = 0; e < 16; ++e) logits[e] += s * gateW[j * 16 + e];
  }
  for (int u = 0; u < 64; ++u) {                // user rows 128..191
    float s = userp[b * 64 + u];
    #pragma unroll
    for (int e = 0; e < 16; ++e) logits[e] += s * gateW[(128 + u) * 16 + e];
  }
  for (int c = 0; c < 16; ++c) {                // context rows 192..207
    float s = ctxf[b * 16 + c];
    #pragma unroll
    for (int e = 0; e < 16; ++e) logits[e] += s * gateW[(192 + c) * 16 + e];
  }
  for (int d = 0; d < 128; ++d) {               // shared rows 208..335
    float s = bf2f(shared_bf[b * 128 + d]);
    #pragma unroll
    for (int e = 0; e < 16; ++e) logits[e] += s * gateW[(208 + d) * 16 + e];
  }

  int i1 = 0; float v1 = logits[0];
  #pragma unroll
  for (int e = 1; e < 16; ++e) if (logits[e] > v1) { v1 = logits[e]; i1 = e; }
  int i2 = -1; float v2 = -3.4e38f;
  #pragma unroll
  for (int e = 0; e < 16; ++e) if (e != i1 && logits[e] > v2) { v2 = logits[e]; i2 = e; }

  float den = 0.f;
  #pragma unroll
  for (int e = 0; e < 16; ++e) den += __expf(logits[e] - v1);
  float g1 = 1.0f / den;
  float g2 = __expf(v2 - v1) / den;
  float rn = 1.0f / (g1 + g2 + 1e-9f);
  float gg1 = g1 * rn, gg2 = g2 * rn;
  #pragma unroll
  for (int e = 0; e < 16; ++e)
    g_out[b * 16 + e] = (e == i1) ? gg1 : ((e == i2) ? gg2 : 0.f);

  float p1 = p_hist[b * 16 + i1], p2 = p_hist[b * 16 + i2];
  float kl = gg1 * (__logf(gg1 + 1e-8f) - __logf(p1 + 1e-8f)) +
             gg2 * (__logf(gg2 + 1e-8f) - __logf(p2 + 1e-8f));

  float* red = smf + 256 * 128;
  red[threadIdx.x] = kl;
  __syncthreads();
  for (int s = 128; s > 0; s >>= 1) {
    if ((int)threadIdx.x < s) red[threadIdx.x] += red[threadIdx.x + s];
    __syncthreads();
  }
  if (threadIdx.x == 0) kl_partial[blockIdx.x] = red[0];
}

// ---------------- K3: fused 16-expert BCN+MLP stack + gated mix + heads ----------------
// Per-wave private LDS (8 x 17408 B) + block-shared double-buffered weight tiles
// (2 x 32 KB) filled with global_load_async_to_lds_b128 one GEMM stage ahead.
#define WAVE_LDS 17408
#define WOFF (8 * WAVE_LDS)                 // 139264: start of weight double buffer
#define WSTAGES (NEXP * 6)                  // 96 weight stages total

__global__ __launch_bounds__(256) void k_experts(
    const u16* __restrict__ shared_bf, const float* __restrict__ g_all,
    const u16* __restrict__ bcnWt, const u16* __restrict__ bcnUt, const float* __restrict__ bcnb,
    const u16* __restrict__ mlpW1t, const float* __restrict__ mlpb1,
    const u16* __restrict__ mlpW2t, const float* __restrict__ mlpb2,
    const float* __restrict__ hctrW, const float* __restrict__ hctrb,
    const float* __restrict__ hcvrW, const float* __restrict__ hcvrb,
    float* __restrict__ ctr_out, float* __restrict__ cvr_out) {
  extern __shared__ char smraw[];
  const int tid = threadIdx.x;
  const int lane = threadIdx.x & 31;
  const int wave = threadIdx.x >> 5;
  const int row0 = (blockIdx.x * 8 + wave) * 16;
  char* base = smraw + wave * WAVE_LDS;
  u16* sbuf = (u16*)base;                       // shared tile, bf16 [16][128]
  u16* abuf = (u16*)(base + 4096);              // A scratch,   bf16 [16][128]
  float* fbuf = (float*)(base + 8192);          // mixed stage, f32  [16][128]
  float* gbuf = (float*)(base + 16384);         // gate tile,   f32  [16][16]

  // weight stage s = e*6 + j: j = {bcnW l0, bcnU l0, bcnW l1, bcnU l1, mlpW1, mlpW2}
  const u16* wseq[6] = {bcnWt, bcnUt, bcnWt, bcnUt, mlpW1t, mlpW2t};
  auto wsrc = [&](int s) -> const u16* {
    int e = s / 6, j = s - e * 6;
    size_t mat = (j < 4) ? (size_t)(e * 2 + (j >> 1)) : (size_t)e;
    return wseq[j] + (mat << 14);
  };

  {
    const uv4* src = (const uv4*)(shared_bf + (size_t)row0 * 128);
    uv4* dst = (uv4*)sbuf;
    #pragma unroll
    for (int i = 0; i < 8; ++i) dst[lane + 32 * i] = src[lane + 32 * i];
    const uv4* gs = (const uv4*)(g_all + (size_t)row0 * 16);
    uv4* gd = (uv4*)gbuf;
    #pragma unroll
    for (int i = 0; i < 2; ++i) gd[lane + 32 * i] = gs[lane + 32 * i];
  }
  lds_fence();

  // prime weight pipeline: stage 0 into buffer 0
  int cur = 0;
  stage_weights_async(wsrc(0), WOFF, tid);
  async_fence();
  __syncthreads();

  auto wlds = [&]() -> const u16* { return (const u16*)(smraw + WOFF + cur * 32768); };
  auto advance = [&]() { async_fence(); __syncthreads(); cur ^= 1; };
  auto prefetch = [&](int s) { if (s < WSTAGES) stage_weights_async(wsrc(s), WOFF + (cur ^ 1) * 32768, tid); };

  const int hh = lane >> 4, nlo = lane & 15;
  v8f mixed[8];
  #pragma unroll
  for (int t = 0; t < 8; ++t) mixed[t] = vzero8();

  for (int e = 0; e < NEXP; ++e) {
    v8f h[8];                                   // f32 residual, C layout
    #pragma unroll
    for (int t = 0; t < 8; ++t)
      #pragma unroll
      for (int v = 0; v < 8; ++v)
        h[t][v] = bf2f(sbuf[(v + 8 * hh) * 128 + t * 16 + nlo]);

    const u16* Abuf = sbuf;
    v8f acc[8];
    #pragma unroll
    for (int l = 0; l < 2; ++l) {               // BCN layers
      prefetch(e * 6 + l * 2 + 1);
      gemm16x128(Abuf, wlds(), acc, lane);
      lds_fence();
      #pragma unroll
      for (int t = 0; t < 8; ++t)               // tmp = (h@W) * h  -> bf16 scratch
        #pragma unroll
        for (int v = 0; v < 8; ++v)
          abuf[(v + 8 * hh) * 128 + t * 16 + nlo] = f2bf(acc[t][v] * h[t][v]);
      lds_fence();
      advance();

      prefetch(e * 6 + l * 2 + 2);
      gemm16x128(abuf, wlds(), acc, lane);
      const float* bb = bcnb + (e * 2 + l) * 128;
      #pragma unroll
      for (int t = 0; t < 8; ++t) {
        float bias = bb[t * 16 + nlo];
        #pragma unroll
        for (int v = 0; v < 8; ++v) {
          float x = acc[t][v] + bias;
          h[t][v] += (x > 0.f) ? x : 0.f;       // h += relu(tmp@U + b)
        }
      }
      lds_fence();
      #pragma unroll
      for (int t = 0; t < 8; ++t)
        #pragma unroll
        for (int v = 0; v < 8; ++v)
          abuf[(v + 8 * hh) * 128 + t * 16 + nlo] = f2bf(h[t][v]);
      lds_fence();
      advance();
      Abuf = abuf;
    }

    prefetch(e * 6 + 5);
    gemm16x128(abuf, wlds(), acc, lane);        // MLP1
    {
      const float* bb = mlpb1 + e * 128;
      lds_fence();
      #pragma unroll
      for (int t = 0; t < 8; ++t) {
        float bias = bb[t * 16 + nlo];
        #pragma unroll
        for (int v = 0; v < 8; ++v) {
          float x = acc[t][v] + bias;
          x = (x > 0.f) ? x : 0.f;
          abuf[(v + 8 * hh) * 128 + t * 16 + nlo] = f2bf(x);
        }
      }
      lds_fence();
    }
    advance();

    prefetch(e * 6 + 6);                        // next expert's first stage
    gemm16x128(abuf, wlds(), acc, lane);        // MLP2 + gated mix
    {
      const float* bb = mlpb2 + e * 128;
      float gv[8];
      #pragma unroll
      for (int v = 0; v < 8; ++v) gv[v] = gbuf[(v + 8 * hh) * 16 + e];
      #pragma unroll
      for (int t = 0; t < 8; ++t) {
        float bias = bb[t * 16 + nlo];
        #pragma unroll
        for (int v = 0; v < 8; ++v) {
          float x = acc[t][v] + bias;
          x = (x > 0.f) ? x : 0.f;
          mixed[t][v] += gv[v] * x;
        }
      }
    }
    advance();
  }

  // heads: stage mixed in f32 LDS, lanes 0-15 do ctr rows, 16-31 do cvr rows
  #pragma unroll
  for (int t = 0; t < 8; ++t)
    #pragma unroll
    for (int v = 0; v < 8; ++v)
      fbuf[(v + 8 * hh) * 128 + t * 16 + nlo] = mixed[t][v];
  lds_fence();
  {
    int r = lane & 15;
    const float* wv = (lane < 16) ? hctrW : hcvrW;
    float s = (lane < 16) ? hctrb[0] : hcvrb[0];
    #pragma unroll 8
    for (int n = 0; n < 128; ++n) s += fbuf[r * 128 + n] * wv[n];
    if (lane < 16) ctr_out[row0 + r] = s;
    else           cvr_out[row0 + r] = s;
  }
}

// ---------------- K4: fixed-order KL reduction -> prior scalar ----------------
__global__ void k_prior(const float* __restrict__ part, int n, float* __restrict__ out) {
  if (threadIdx.x == 0 && blockIdx.x == 0) {
    float s = 0.f;
    for (int i = 0; i < n; ++i) s += part[i];
    *out = s / (float)B_ROWS;
  }
}

extern "C" void kernel_launch(void* const* d_in, const int* in_sizes, int n_in,
                              void* d_out, int out_size, void* d_ws, size_t ws_size,
                              hipStream_t stream) {
  (void)in_sizes; (void)n_in; (void)out_size; (void)ws_size;

  const int* ids0 = (const int*)d_in[0];
  const int* ids1 = (const int*)d_in[1];
  const int* ids2 = (const int*)d_in[2];
  const int* ids3 = (const int*)d_in[3];
  const int* resource_id = (const int*)d_in[4];
  /* d_in[5] hist_scene_dist: zero-width, unused */
  const float* realtime = (const float*)d_in[6];
  const float* userp = (const float*)d_in[7];
  const float* ctxf = (const float*)d_in[8];
  const float* p_hist = (const float*)d_in[9];
  /* d_in[10] topk == 2, hardcoded */
  const float* emb0 = (const float*)d_in[11];
  const float* emb1 = (const float*)d_in[12];
  const float* emb2 = (const float*)d_in[13];
  const float* emb3 = (const float*)d_in[14];
  const float* proj_W = (const float*)d_in[15];
  const float* proj_b = (const float*)d_in[16];
  const float* res_emb = (const float*)d_in[17];
  const float* sc_W1 = (const float*)d_in[18];
  const float* sc_b1 = (const float*)d_in[19];
  const float* sc_W2 = (const float*)d_in[20];
  const float* sc_b2 = (const float*)d_in[21];
  const float* gate_W = (const float*)d_in[22];
  const float* gate_b = (const float*)d_in[23];
  const float* bcn_W = (const float*)d_in[24];
  const float* bcn_U = (const float*)d_in[25];
  const float* bcn_b = (const float*)d_in[26];
  const float* mlp_W1 = (const float*)d_in[27];
  const float* mlp_b1 = (const float*)d_in[28];
  const float* mlp_W2 = (const float*)d_in[29];
  const float* mlp_b2 = (const float*)d_in[30];
  const float* hctrW = (const float*)d_in[31];
  const float* hctrb = (const float*)d_in[32];
  const float* hcvrW = (const float*)d_in[33];
  const float* hcvrb = (const float*)d_in[34];

  char* ws = (char*)d_ws;
  size_t off = 0;
  u16* projWt = (u16*)(ws + off); off += (size_t)512 * 128 * 2;        // 128 KB
  u16* bcnWt  = (u16*)(ws + off); off += (size_t)32 * 128 * 128 * 2;   // 1 MB
  u16* bcnUt  = (u16*)(ws + off); off += (size_t)32 * 128 * 128 * 2;   // 1 MB
  u16* mlpW1t = (u16*)(ws + off); off += (size_t)16 * 128 * 128 * 2;   // 512 KB
  u16* mlpW2t = (u16*)(ws + off); off += (size_t)16 * 128 * 128 * 2;   // 512 KB
  u16* sharedBf = (u16*)(ws + off); off += (size_t)B_ROWS * 128 * 2;   // 4 MB
  float* klPart = (float*)(ws + off); off += 64 * 4;

  float* out = (float*)d_out;
  float* ctr_out = out;
  float* cvr_out = out + B_ROWS;
  float* g_out = out + 2 * B_ROWS;
  float* prior_out = out + 2 * B_ROWS + B_ROWS * NEXP;

  // 0) weights -> bf16 W^T
  k_conv_t<<<(512 * 128 + 255) / 256, 256, 0, stream>>>(proj_W, projWt, 512, 128, 512 * 128);
  k_conv_t<<<(32 * 16384 + 255) / 256, 256, 0, stream>>>(bcn_W, bcnWt, 128, 128, 32 * 16384);
  k_conv_t<<<(32 * 16384 + 255) / 256, 256, 0, stream>>>(bcn_U, bcnUt, 128, 128, 32 * 16384);
  k_conv_t<<<(16 * 16384 + 255) / 256, 256, 0, stream>>>(mlp_W1, mlpW1t, 128, 128, 16 * 16384);
  k_conv_t<<<(16 * 16384 + 255) / 256, 256, 0, stream>>>(mlp_W2, mlpW2t, 128, 128, 16 * 16384);

  // 1) shared bottom (WMMA, K=512), 8 waves/block, 128 rows/block
  k_shared_bottom<<<B_ROWS / 128, 256, 8 * 16 * 512 * (int)sizeof(u16), stream>>>(
      ids0, ids1, ids2, ids3, emb0, emb1, emb2, emb3, projWt, proj_b, sharedBf);

  // 2) scene + gate + KL partials (writes g into d_out)
  k_gate<<<B_ROWS / 256, 256, (256 * 128 + 256) * (int)sizeof(float), stream>>>(
      resource_id, realtime, userp, ctxf, p_hist, sharedBf,
      res_emb, sc_W1, sc_b1, sc_W2, sc_b2, gate_W, gate_b, g_out, klPart);

  // 3) fused expert stack + gated mix + heads
  //    LDS: 8 x 17408 (wave-private) + 2 x 32768 (async-staged weight double buffer)
  k_experts<<<B_ROWS / 128, 256, WOFF + 2 * 32768, stream>>>(
      sharedBf, g_out, bcnWt, bcnUt, bcn_b, mlpW1t, mlp_b1, mlpW2t, mlp_b2,
      hctrW, hctrb, hcvrW, hcvrb, ctr_out, cvr_out);

  // 4) prior scalar (deterministic fixed-order sum)
  k_prior<<<1, 32, 0, stream>>>(klPart, B_ROWS / 256, prior_out);
}
